// LSTM_80470507257935
// MI455X (gfx1250) — matrix operand order repaired and checked by reading, not compiled, not run on hardware
//
#include <hip/hip_runtime.h>
#include <hip/hip_bf16.h>

typedef __attribute__((ext_vector_type(16))) __bf16 v16bf;
typedef __attribute__((ext_vector_type(8)))  __bf16 v8bf;
typedef __attribute__((ext_vector_type(8)))  float  v8f;

#define DIN   64      // input dim
#define HID   64      // hidden dim
#define KDIM  128     // D + H
#define NDIM  256     // 4*H
#define BM    64      // events per block

// ---------------- weight prep: fold W_ih|W_hh -> bf16, b_ih+b_hh -> f32 ----
__global__ void lstm_prep_kernel(const float* __restrict__ Wih,
                                 const float* __restrict__ Whh,
                                 const float* __restrict__ bih,
                                 const float* __restrict__ bhh,
                                 __bf16* __restrict__ Wc,   // [256][128]
                                 float*  __restrict__ bc) { // [256]
  int t = threadIdx.x + blockIdx.x * blockDim.x;
  int stride = blockDim.x * gridDim.x;
  for (int p = t; p < NDIM * KDIM; p += stride) {
    int n = p >> 7;
    int k = p & 127;
    float v = (k < DIN) ? Wih[n * DIN + k] : Whh[n * HID + (k - DIN)];
    Wc[p] = (__bf16)v;
  }
  for (int n = t; n < NDIM; n += stride) bc[n] = bih[n] + bhh[n];
}

// ---------------- fused gather + gates GEMM (WMMA) + LSTM elementwise ------
__global__ __launch_bounds__(256) void lstm_fused_kernel(
    const float* __restrict__ x,       // [E, 64]
    const int*   __restrict__ ids,     // [E]
    const float* __restrict__ hmem,    // [N_NODES, 64]
    const float* __restrict__ cmem,    // [N_NODES, 64]
    const __bf16* __restrict__ Wc,     // [256][128] bf16 (row n = gate row, k contiguous)
    const float* __restrict__ bc,      // [256]
    const float* __restrict__ linw,    // [64]
    const float* __restrict__ linb,    // [1]
    float* __restrict__ out,           // [E]
    int E) {

  __shared__ __bf16 sA[BM * KDIM];   // [e][k] : x | h  (bf16 A panel)
  __shared__ float  sC[BM * HID];    // gathered cell state
  __shared__ float  sG[BM * NDIM];   // f32 gates (pre-bias)
  __shared__ float  sB[NDIM];        // combined bias
  __shared__ int    sIds[BM];

  const int tid  = threadIdx.x;
  const int base = blockIdx.x * BM;

  sB[tid] = bc[tid];                 // 256 threads, 256 biases
  if (tid < BM) {
    int ge = base + tid;
    sIds[tid] = (ge < E) ? ids[ge] : 0;
  }
  __syncthreads();

  // ---- Phase 1: gather + f32->bf16 convert into LDS (coalesced) ----
  #pragma unroll
  for (int it = 0; it < (BM * HID) / 256; ++it) {
    int p = it * 256 + tid;
    int e = p >> 6;          // local event
    int k = p & 63;          // column
    int ge = base + e;
    int id = sIds[e];
    float xv = (ge < E) ? x[(size_t)ge * DIN + k] : 0.f;
    float hv = hmem[(size_t)id * HID + k];
    float cv = cmem[(size_t)id * HID + k];
    sA[e * KDIM + k]        = (__bf16)xv;
    sA[e * KDIM + DIN + k]  = (__bf16)hv;
    sC[e * HID + k]         = cv;
  }
  __syncthreads();

  // ---- Phase 2: gates[64][256] = A[64][128] * Wc^T via WMMA bf16 ----
  const int wave = tid >> 5;
  const int lane = tid & 31;
  const int l16  = lane & 15;
  const int hi   = lane >> 4;        // 0: lanes 0-15, 1: lanes 16-31
  const int mt   = wave & 3;         // M tile (4 x 16 rows)
  const int nt0  = (wave >> 2) * 8;  // this wave covers 8 N tiles

  // A fragments per ISA 16-bit A 16x32 layout:
  //   lanes 0-15: halves 0-7 = K kb..kb+7,  halves 8-15 = K kb+16..kb+23, kb=0
  //   lanes16-31: same with kb=8
  v16bf afrag[4];
  {
    const __bf16* ar = &sA[(mt * 16 + l16) * KDIM];
    const int kb = hi * 8;
    #pragma unroll
    for (int f = 0; f < 4; ++f) {
      *(v8bf*)&afrag[f]       = *(const v8bf*)(ar + kb + f * 32);
      *((v8bf*)&afrag[f] + 1) = *(const v8bf*)(ar + kb + f * 32 + 16);
    }
  }

  #pragma unroll
  for (int ntl = 0; ntl < 8; ++ntl) {
    int nt = nt0 + ntl;
    int n  = nt * 16 + l16;                 // gate column this lane owns
    v8f acc = {};                           // inline-0 C for first WMMA (no mov flood)
    // B 32x16 layout: lane col = l16; lanes 0-15 hold K 0-7,16-23, lanes 16-31 hold K 8-15,24-31.
    // B[k][n] = Wc[n][k] -> two runs of 8 contiguous bf16 from row n.
    const __bf16* br = Wc + (size_t)n * KDIM + hi * 8;
    #pragma unroll
    for (int f = 0; f < 4; ++f) {
      v16bf bfrag;
      *(v8bf*)&bfrag       = *(const v8bf*)(br + f * 32);
      *((v8bf*)&bfrag + 1) = *(const v8bf*)(br + f * 32 + 16);
      acc = __builtin_amdgcn_wmma_f32_16x16x32_bf16(
          /*neg_a=*/false, afrag[f], /*neg_b=*/false, bfrag,
          /*c_mod=*/(short)0, acc, /*reuse_a=*/false, /*reuse_b=*/false);
    }
    // C/D layout: VGPR r -> row (mt*16 + hi*8 + r), col n
    int m = mt * 16 + hi * 8;
    #pragma unroll
    for (int r = 0; r < 8; ++r) sG[(m + r) * NDIM + n] = acc[r];
  }
  __syncthreads();

  // ---- Phase 3: bias + LSTM elementwise + readout dot ----
  // 4 lanes per event, 16 hidden channels each, shfl reduction.
  const float lb = linb[0];
  const int e  = tid >> 2;
  const int j0 = (tid & 3) * 16;
  float sum = 0.f;
  #pragma unroll
  for (int jj = 0; jj < 16; ++jj) {
    int j = j0 + jj;
    float ig = sG[e * NDIM + j]        + sB[j];
    float fg = sG[e * NDIM + 64  + j]  + sB[64 + j];
    float gg = sG[e * NDIM + 128 + j]  + sB[128 + j];
    float og = sG[e * NDIM + 192 + j]  + sB[192 + j];
    float cv = sC[e * HID + j];
    float si = 1.f / (1.f + __expf(-ig));
    float sf = 1.f / (1.f + __expf(-fg));
    float so = 1.f / (1.f + __expf(-og));
    float cn = sf * cv + si * tanhf(gg);
    float hn = so * tanhf(cn);
    sum += hn * linw[j];
  }
  sum += __shfl_xor(sum, 1, 32);
  sum += __shfl_xor(sum, 2, 32);
  if ((tid & 3) == 0) {
    int ge = base + e;
    if (ge < E) out[ge] = sum + lb;
  }
}

extern "C" void kernel_launch(void* const* d_in, const int* in_sizes, int n_in,
                              void* d_out, int out_size, void* d_ws, size_t ws_size,
                              hipStream_t stream) {
  const float* x    = (const float*)d_in[0];  // node_embs [E,64]
  const int*   ids  = (const int*)  d_in[1];  // dst_node_ids [E]
  const float* hmem = (const float*)d_in[2];  // h_mem
  const float* cmem = (const float*)d_in[3];  // c_mem
  const float* Wih  = (const float*)d_in[4];
  const float* Whh  = (const float*)d_in[5];
  const float* bih  = (const float*)d_in[6];
  const float* bhh  = (const float*)d_in[7];
  const float* linw = (const float*)d_in[8];
  const float* linb = (const float*)d_in[9];
  float* out = (float*)d_out;

  const int E = in_sizes[1];

  __bf16* Wc  = (__bf16*)d_ws;
  float*  bcm = (float*)((char*)d_ws + (size_t)NDIM * KDIM * sizeof(__bf16));

  lstm_prep_kernel<<<32, 256, 0, stream>>>(Wih, Whh, bih, bhh, Wc, bcm);

  int blocks = (E + BM - 1) / BM;
  lstm_fused_kernel<<<blocks, 256, 0, stream>>>(x, ids, hmem, cmem, Wc, bcm,
                                                linw, linb, out, E);
}
